// EvaAttention_2997887172752
// MI455X (gfx1250) — matrix operand order repaired and checked by reading, not compiled, not run on hardware
//
#include <hip/hip_runtime.h>
#include <stdint.h>

// ---------------------------------------------------------------------------
// MI455X (gfx1250) attention block, wave32 WMMA bf16 (f32 accum), with
// CDNA5 async global->LDS staging (ASYNCcnt, saddr+imm form) and LDS
// double buffering.
// Fragment layouts per CDNA5 ISA 7.12.2:
//   A (16x32 bf16): lane m = lane%16; halves: K=(lane<16?0:8)+0..7 then
//                   K=16+(lane<16?0:8)+0..7  -> 2x b128 per fragment
//   B (32x16 bf16): lane n = lane%16; K order as A; this code uses K=(lane<16?0:16)+0..15 for A and B
//   C/D (16x16 f32): VGPR r -> m = r + (lane<16?0:8), n = lane%16
// ---------------------------------------------------------------------------

typedef __attribute__((ext_vector_type(16))) __bf16 v16bf;
typedef __attribute__((ext_vector_type(8)))  float  v8f;

union Frag {
  v16bf v;
  uint4 u[2];
};

__device__ __forceinline__ v8f zero8() {
  v8f z;
#pragma unroll
  for (int i = 0; i < 8; ++i) z[i] = 0.0f;
  return z;
}

// xor-shuffle within 16-lane halves via ds_swizzle (imm pattern, no index reg)
template <int M>
__device__ __forceinline__ float swz_xor(float v) {
  return __int_as_float(
      __builtin_amdgcn_ds_swizzle(__float_as_int(v), 0x1f | (M << 10)));
}

// Generic pointer into LDS carries the LDS byte offset in its low 32 bits
// (ISA 10.2, LDS aperture).
__device__ __forceinline__ uint32_t lds_addr(const void* p) {
  return (uint32_t)(uintptr_t)p;
}

// 16B async copy global->LDS, saddr form: mem = SADDR + VOFF + OFF,
// lds = VDST + OFF (ISA 8: INST_OFFSET applies to both sides).
template <int OFF>
__device__ __forceinline__ void async_cp16s(uint32_t lds, uint32_t voff,
                                            const void* sbase) {
  asm volatile("global_load_async_to_lds_b128 %0, %1, %2 offset:%c3"
               :: "v"(lds), "v"(voff), "s"((uint64_t)(uintptr_t)sbase),
                  "n"(OFF)
               : "memory");
}

__device__ __forceinline__ void wait_async0() {
  asm volatile("s_wait_asynccnt 0x0" ::: "memory");
}

// ---------------------------------------------------------------------------
// f32 -> bf16 bulk convert (4 elems/thread)
// ---------------------------------------------------------------------------
__global__ __launch_bounds__(256) void convert_kernel(
    const float* __restrict__ src, __bf16* __restrict__ dst) {
  int i = (blockIdx.x * 256 + threadIdx.x) * 4;
  float4 f = *(const float4*)(src + i);
  union { uint2 u; __bf16 h[4]; } cv;
  cv.h[0] = (__bf16)f.x; cv.h[1] = (__bf16)f.y;
  cv.h[2] = (__bf16)f.z; cv.h[3] = (__bf16)f.w;
  *(uint2*)(dst + i) = cv.u;
}

// bias concat [q_bias | zeros | v_bias] (3072 f32)
__global__ __launch_bounds__(256) void build_bias_kernel(
    const float* __restrict__ qb, const float* __restrict__ vb,
    float* __restrict__ out) {
  int i = blockIdx.x * 256 + threadIdx.x;
  float v = 0.0f;
  if (i < 1024)       v = qb[i];
  else if (i >= 2048) v = vb[i - 2048];
  out[i] = v;
}

// ---------------------------------------------------------------------------
// Shared GEMM mainloop, async double-buffered, k-loop unrolled by 2 so the
// buffer index is compile-time. acc[2][4] += A_tile(128x32)*W_tile(128x32)^T.
// ---------------------------------------------------------------------------
#define TM 128
#define TN 128
#define TK 32
#define LDT 40          // padded LDS row stride (halves); 80B rows, 16B aligned
#define TILE (TM * LDT) // halves per buffer

template <int K>
__device__ __forceinline__ void gemm_mainloop(
    const __bf16* __restrict__ A, const __bf16* __restrict__ W,
    int m0, int n0, __bf16* Abuf, __bf16* Bbuf, v8f acc[2][4]) {
  const int tid   = threadIdx.x;
  const int lane  = tid & 31;
  const int wave  = tid >> 5;
  const int lhalf = lane >> 4;
  const int l16   = lane & 15;
  const int wm = wave & 3;
  const int wn = wave >> 2;

#pragma unroll
  for (int i = 0; i < 2; ++i)
#pragma unroll
    for (int j = 0; j < 4; ++j) acc[i][j] = zero8();

  // this thread's two 16B chunks: same row, kq and kq+8 halves (+16 B)
  const int crow = (tid * 2) >> 2;        // 0..127
  const int ckq  = ((tid * 2) & 3) * 8;   // 0 or 16 (halves)

  const __bf16* sA = A + (size_t)m0 * K;  // uniform SGPR bases
  const __bf16* sB = W + (size_t)n0 * K;
  uint32_t voff = (uint32_t)((crow * K + ckq) * 2);   // shared A/B voffset

  uint32_t ldsA0 = lds_addr(&Abuf[crow * LDT + ckq]);
  uint32_t ldsB0 = lds_addr(&Bbuf[crow * LDT + ckq]);

  auto stage = [&](uint32_t dA, uint32_t dB) {
    async_cp16s<0>(dA, voff, sA);
    async_cp16s<16>(dA, voff, sA);
    async_cp16s<0>(dB, voff, sB);
    async_cp16s<16>(dB, voff, sB);
    voff += TK * 2;   // advance one k-tile (bytes)
  };

  auto compute = [&](const __bf16* Ab, const __bf16* Bb) {
    Frag af[2], bf[4];
#pragma unroll
    for (int ms = 0; ms < 2; ++ms) {
      const __bf16* p = &Ab[(wm * 32 + ms * 16 + l16) * LDT + lhalf * 16];
      af[ms].u[0] = *(const uint4*)p;
      af[ms].u[1] = *(const uint4*)(p + 8);
    }
#pragma unroll
    for (int ns = 0; ns < 4; ++ns) {
      const __bf16* p = &Bb[(wn * 64 + ns * 16 + l16) * LDT + lhalf * 16];
      bf[ns].u[0] = *(const uint4*)p;
      bf[ns].u[1] = *(const uint4*)(p + 8);
    }
#pragma unroll
    for (int ms = 0; ms < 2; ++ms)
#pragma unroll
      for (int ns = 0; ns < 4; ++ns)
        acc[ms][ns] = __builtin_amdgcn_wmma_f32_16x16x32_bf16(
            false, af[ms].v, false, bf[ns].v, (short)0, acc[ms][ns],
            false, false);
  };

  stage(ldsA0, ldsB0);
  wait_async0();
  __syncthreads();

  const int nIter = K / TK;   // even
  for (int it = 0; it < nIter; it += 2) {
    // buffer 0 resident: prefetch into buffer 1, compute from buffer 0
    stage(ldsA0 + TILE * 2, ldsB0 + TILE * 2);
    compute(&Abuf[0], &Bbuf[0]);
    wait_async0();
    __syncthreads();
    // buffer 1 resident: prefetch into buffer 0, compute from buffer 1
    if (it + 2 < nIter) stage(ldsA0, ldsB0);
    compute(&Abuf[TILE], &Bbuf[TILE]);
    wait_async0();
    __syncthreads();
  }
}

// ---------------------------------------------------------------------------
// QKV GEMM with fused bias + RoPE + split to q/k/v bf16 [B,H,S,D].
// A wave's 64 N-columns are exactly one (part, head) with d = 0..63, and the
// RoPE partner (d^32) is acc[ms][ns^2][r] in the SAME lane -> register-only.
// ---------------------------------------------------------------------------
__global__ __launch_bounds__(256) void qkv_gemm_kernel(
    const __bf16* __restrict__ A, const __bf16* __restrict__ W,
    const float* __restrict__ bias, const float* __restrict__ rsin,
    const float* __restrict__ rcos, __bf16* __restrict__ qo,
    __bf16* __restrict__ ko, __bf16* __restrict__ vo) {
  __shared__ __align__(16) __bf16 Abuf[2 * TILE];
  __shared__ __align__(16) __bf16 Bbuf[2 * TILE];
  const int S = 2048, H = 16, D = 64;
  const int n0 = blockIdx.x * TN;
  const int m0 = blockIdx.y * TM;
  const int lane  = threadIdx.x & 31;
  const int wave  = threadIdx.x >> 5;
  const int lhalf = lane >> 4;
  const int l16   = lane & 15;
  const int wm = wave & 3;
  const int wn = wave >> 2;

  v8f acc[2][4];
  gemm_mainloop<1024>(A, W, m0, n0, Abuf, Bbuf, acc);

  const int gnb  = n0 + wn * 64;       // multiple of 64
  const int part = gnb >> 10;          // 0=q 1=k 2=v
  const int hh   = (gnb >> 6) & 15;
  __bf16* outp = (part == 0) ? qo : (part == 1) ? ko : vo;

  float bv[4];
#pragma unroll
  for (int ns = 0; ns < 4; ++ns) bv[ns] = bias[gnb + ns * 16 + l16];

#pragma unroll
  for (int ms = 0; ms < 2; ++ms) {
    int mbase = m0 + wm * 32 + ms * 16 + 8 * lhalf;
#pragma unroll
    for (int r = 0; r < 8; ++r) {
      int m = mbase + r;
      int s = m & (S - 1);
      int bb = m >> 11;
      bool dorope = (part < 2) && (s >= 1);
      size_t obase = ((size_t)(bb * H + hh) * S + s) * D;
#pragma unroll
      for (int ns = 0; ns < 4; ++ns) {
        int d = ns * 16 + l16;
        float t = acc[ms][ns][r] + bv[ns];
        if (dorope) {
          float tp  = acc[ms][ns ^ 2][r] + bv[ns ^ 2];
          float rot = (ns < 2) ? -tp : tp;
          float cs  = rcos[(size_t)(s - 1) * D + d];
          float sn  = rsin[(size_t)(s - 1) * D + d];
          t = t * cs + rot * sn;
        }
        outp[obase + d] = (__bf16)t;
      }
    }
  }
}

// ---------------------------------------------------------------------------
// Projection GEMM: Out f32 = A * W^T + bias
// ---------------------------------------------------------------------------
__global__ __launch_bounds__(256) void proj_gemm_kernel(
    const __bf16* __restrict__ A, const __bf16* __restrict__ W,
    const float* __restrict__ bias, float* __restrict__ Out, int N) {
  __shared__ __align__(16) __bf16 Abuf[2 * TILE];
  __shared__ __align__(16) __bf16 Bbuf[2 * TILE];
  const int n0 = blockIdx.x * TN;
  const int m0 = blockIdx.y * TM;
  const int lane  = threadIdx.x & 31;
  const int wave  = threadIdx.x >> 5;
  const int lhalf = lane >> 4;
  const int l16   = lane & 15;
  const int wm = wave & 3;
  const int wn = wave >> 2;

  v8f acc[2][4];
  gemm_mainloop<1024>(A, W, m0, n0, Abuf, Bbuf, acc);

#pragma unroll
  for (int ms = 0; ms < 2; ++ms) {
    int gm = m0 + wm * 32 + ms * 16 + 8 * lhalf;
#pragma unroll
    for (int ns = 0; ns < 4; ++ns) {
      int gn = n0 + wn * 64 + ns * 16 + l16;
      float bvv = bias[gn];
      float* o = Out + (size_t)gm * N + gn;
#pragma unroll
      for (int r = 0; r < 8; ++r)
        o[(size_t)r * N] = acc[ms][ns][r] + bvv;
    }
  }
}

// ---------------------------------------------------------------------------
// Flash attention per (b,h): 128 q rows/block (16 per wave), key blocks of 64.
// K staged via async copies (saddr+imm); V transposed 4x4 in registers.
// Online softmax with ds_swizzle xor-reductions. Writes bf16 [B,S,C].
// ---------------------------------------------------------------------------
__global__ __launch_bounds__(256) void attn_kernel(
    const __bf16* __restrict__ Q, const __bf16* __restrict__ Kb,
    const __bf16* __restrict__ V, __bf16* __restrict__ Ob) {
  const int S = 2048, D = 64;
  const float scale = 0.125f;
  __shared__ __align__(16) __bf16 Kbuf[64 * 72];     // K block [key][d]
  __shared__ __align__(16) __bf16 Vt[64 * 72];       // V^T block [d][key]
  __shared__ __align__(16) __bf16 Pb[8 * 16 * 72];   // per-wave P slab

  int tid = threadIdx.x;
  int lane = tid & 31, wave = tid >> 5;
  int lhalf = lane >> 4, l16 = lane & 15;
  int bh = blockIdx.y;
  int b = bh >> 4, h = bh & 15;
  int q0 = blockIdx.x * 128 + wave * 16;

  const __bf16* qg = Q + (size_t)bh * S * D;
  const __bf16* kg = Kb + (size_t)bh * S * D;
  const __bf16* vg = V + (size_t)bh * S * D;

  // resident Q A-fragments (2 K-chunks over D=64)
  Frag aq[2];
  {
    const __bf16* qrow = qg + (size_t)(q0 + l16) * D;
#pragma unroll
    for (int c = 0; c < 2; ++c) {
      aq[c].u[0] = *(const uint4*)(qrow + 32 * c + lhalf * 16);
      aq[c].u[1] = *(const uint4*)(qrow + 32 * c + 8 + lhalf * 16);
    }
  }

  float m_i[8], l_i[8];
  v8f oacc[4];
#pragma unroll
  for (int r = 0; r < 8; ++r) { m_i[r] = -1e30f; l_i[r] = 0.0f; }
#pragma unroll
  for (int t = 0; t < 4; ++t) oacc[t] = zero8();

  __bf16* Pl = &Pb[wave * 16 * 72];
  const int kgrp = tid & 15;   // 4-key group for V transpose
  const int dgrp = tid >> 4;   // 4-d group

  // K staging: two 16B chunks per thread, same row, +16B apart
  const int krow = (tid * 2) >> 3;
  const int kkq  = ((tid * 2) & 7) * 8;
  uint32_t kvoff = (uint32_t)((krow * D + kkq) * 2);
  const uint32_t ldsK = lds_addr(&Kbuf[krow * 72 + kkq]);

  for (int kb = 0; kb < S; kb += 64) {
    __syncthreads();
    async_cp16s<0>(ldsK, kvoff, kg);
    async_cp16s<16>(ldsK, kvoff, kg);
    kvoff += 64 * D * 2;   // next key block
    // V: register 4x4 transpose -> Vt[d][key]
    {
      union { uint2 u; __bf16 h4[4]; } in[4], ot[4];
#pragma unroll
      for (int j = 0; j < 4; ++j)
        in[j].u = *(const uint2*)(vg + (size_t)(kb + kgrp * 4 + j) * D + dgrp * 4);
#pragma unroll
      for (int j = 0; j < 4; ++j)
#pragma unroll
        for (int i = 0; i < 4; ++i) ot[j].h4[i] = in[i].h4[j];
#pragma unroll
      for (int j = 0; j < 4; ++j)
        *(uint2*)&Vt[(dgrp * 4 + j) * 72 + kgrp * 4] = ot[j].u;
    }
    wait_async0();
    __syncthreads();

    // logits = Q K^T
    v8f lacc[4];
#pragma unroll
    for (int nt = 0; nt < 4; ++nt) {
      lacc[nt] = zero8();
      const __bf16* krowp = &Kbuf[(nt * 16 + l16) * 72];
      Frag bk0, bk1;
      bk0.u[0] = *(const uint4*)(krowp + lhalf * 16);
      bk0.u[1] = *(const uint4*)(krowp + lhalf * 16 + 8);
      bk1.u[0] = *(const uint4*)(krowp + 32 + lhalf * 16);
      bk1.u[1] = *(const uint4*)(krowp + 32 + lhalf * 16 + 8);
      lacc[nt] = __builtin_amdgcn_wmma_f32_16x16x32_bf16(
          false, aq[0].v, false, bk0.v, (short)0, lacc[nt], false, false);
      lacc[nt] = __builtin_amdgcn_wmma_f32_16x16x32_bf16(
          false, aq[1].v, false, bk1.v, (short)0, lacc[nt], false, false);
    }

    // online softmax; row r lives in one 16-lane half (column = l16)
#pragma unroll
    for (int r = 0; r < 8; ++r) {
      float x0 = lacc[0][r] * scale, x1 = lacc[1][r] * scale;
      float x2 = lacc[2][r] * scale, x3 = lacc[3][r] * scale;
      float mx = fmaxf(fmaxf(x0, x1), fmaxf(x2, x3));
      mx = fmaxf(mx, swz_xor<1>(mx));
      mx = fmaxf(mx, swz_xor<2>(mx));
      mx = fmaxf(mx, swz_xor<4>(mx));
      mx = fmaxf(mx, swz_xor<8>(mx));
      float mnew  = fmaxf(m_i[r], mx);
      float alpha = __expf(m_i[r] - mnew);
      float p0 = __expf(x0 - mnew), p1 = __expf(x1 - mnew);
      float p2 = __expf(x2 - mnew), p3 = __expf(x3 - mnew);
      float ps = p0 + p1 + p2 + p3;
      ps += swz_xor<1>(ps);
      ps += swz_xor<2>(ps);
      ps += swz_xor<4>(ps);
      ps += swz_xor<8>(ps);
      l_i[r] = l_i[r] * alpha + ps;
      m_i[r] = mnew;
#pragma unroll
      for (int dt = 0; dt < 4; ++dt) oacc[dt][r] *= alpha;
      int rr = r + 8 * lhalf;
      Pl[rr * 72 +  0 + l16] = (__bf16)p0;
      Pl[rr * 72 + 16 + l16] = (__bf16)p1;
      Pl[rr * 72 + 32 + l16] = (__bf16)p2;
      Pl[rr * 72 + 48 + l16] = (__bf16)p3;
    }

    // PV
    Frag ap[2];
#pragma unroll
    for (int c = 0; c < 2; ++c) {
      const __bf16* pr = Pl + l16 * 72 + 32 * c + lhalf * 16;
      ap[c].u[0] = *(const uint4*)pr;
      ap[c].u[1] = *(const uint4*)(pr + 8);
    }
#pragma unroll
    for (int dt = 0; dt < 4; ++dt) {
      const __bf16* vr = &Vt[(dt * 16 + l16) * 72];
      Frag bv0, bv1;
      bv0.u[0] = *(const uint4*)(vr + lhalf * 16);
      bv0.u[1] = *(const uint4*)(vr + lhalf * 16 + 8);
      bv1.u[0] = *(const uint4*)(vr + 32 + lhalf * 16);
      bv1.u[1] = *(const uint4*)(vr + 32 + lhalf * 16 + 8);
      oacc[dt] = __builtin_amdgcn_wmma_f32_16x16x32_bf16(
          false, ap[0].v, false, bv0.v, (short)0, oacc[dt], false, false);
      oacc[dt] = __builtin_amdgcn_wmma_f32_16x16x32_bf16(
          false, ap[1].v, false, bv1.v, (short)0, oacc[dt], false, false);
    }
  }

  // epilogue: O / l, write bf16 [B,S,C], C column = h*64 + d
#pragma unroll
  for (int dt = 0; dt < 4; ++dt) {
#pragma unroll
    for (int r = 0; r < 8; ++r) {
      int row = q0 + r + 8 * lhalf;
      int col = h * 64 + dt * 16 + l16;
      float v = oacc[dt][r] / l_i[r];
      Ob[(size_t)(b * S + row) * 1024 + col] = (__bf16)v;
    }
  }
}

// ---------------------------------------------------------------------------
// Host-side launch
// ---------------------------------------------------------------------------
extern "C" void kernel_launch(void* const* d_in, const int* in_sizes, int n_in,
                              void* d_out, int out_size, void* d_ws,
                              size_t ws_size, hipStream_t stream) {
  (void)in_sizes; (void)n_in; (void)out_size; (void)ws_size;
  const float* x      = (const float*)d_in[0];
  const float* qkv_w  = (const float*)d_in[1];
  const float* q_bias = (const float*)d_in[2];
  const float* v_bias = (const float*)d_in[3];
  const float* proj_w = (const float*)d_in[4];
  const float* proj_b = (const float*)d_in[5];
  const float* rsin   = (const float*)d_in[6];
  const float* rcos   = (const float*)d_in[7];
  float* out = (float*)d_out;

  const int B = 4, S = 2048, C = 1024, H = 16;
  const size_t M = (size_t)B * S;   // 8192

  char* ws = (char*)d_ws;
  __bf16* xbf  = (__bf16*)ws;  ws += M * C * 2;             // 16 MB
  __bf16* wq   = (__bf16*)ws;  ws += (size_t)3 * C * C * 2; // 6 MB
  __bf16* wp   = (__bf16*)ws;  ws += (size_t)C * C * 2;     // 2 MB
  __bf16* qb   = (__bf16*)ws;  ws += M * C * 2;
  __bf16* kb   = (__bf16*)ws;  ws += M * C * 2;
  __bf16* vb   = (__bf16*)ws;  ws += M * C * 2;
  __bf16* ab   = (__bf16*)ws;  ws += M * C * 2;
  float*  qkvb = (float*)ws;   ws += 3 * C * sizeof(float);

  convert_kernel<<<dim3((unsigned)(M * C / 1024)), 256, 0, stream>>>(x, xbf);
  convert_kernel<<<dim3(3 * C * C / 1024), 256, 0, stream>>>(qkv_w, wq);
  convert_kernel<<<dim3(C * C / 1024), 256, 0, stream>>>(proj_w, wp);
  build_bias_kernel<<<dim3(3 * C / 256), 256, 0, stream>>>(q_bias, v_bias, qkvb);

  qkv_gemm_kernel<<<dim3(3 * C / TN, M / TM), 256, 0, stream>>>(
      xbf, wq, qkvb, rsin, rcos, qb, kb, vb);

  attn_kernel<<<dim3(S / 128, B * H), 256, 0, stream>>>(qb, kb, vb, ab);

  proj_gemm_kernel<<<dim3(C / TN, M / TM), 256, 0, stream>>>(
      ab, wp, proj_b, out, C);
}
